// FFTConvModule_26774826123742
// MI455X (gfx1250) — matrix operand order repaired and checked by reading, not compiled, not run on hardware
//
#include <hip/hip_runtime.h>

typedef _Float16 f16;
typedef __attribute__((ext_vector_type(16))) _Float16 v16h;
typedef __attribute__((ext_vector_type(8)))  float    v8f;
typedef int v4i __attribute__((__vector_size__(16)));

#define AS1 __attribute__((address_space(1)))
#define AS3 __attribute__((address_space(3)))

namespace {
constexpr int Bn      = 4;
constexpr int Dn      = 1024;
constexpr int Ln      = 4096;
constexpr int WIN     = 16;   // k-pair steps staged per window
constexpr int NSTAGES = 8;    // 8*16 = 128 steps -> lags k in [0,255]
constexpr int BLOCK   = 256;  // 8 waves (wave32)
}

// Causal long conv as banded block-Toeplitz matmul on the f16 WMMA pipe,
// f32 accumulate, 2-term f16 error-split (hi*hi + hi*lo + lo*hi).
__global__ __launch_bounds__(BLOCK)
void fftconv_lag_wmma(const float* __restrict__ xg,
                      const float* __restrict__ fg,
                      float* __restrict__ yg)
{
    // LDS: 16KB filter(f32) + 2*32KB x hi/lo (all 4 batches) + 2*16KB A-frags
    __shared__ __attribute__((aligned(32))) float hbuf[Ln];
    __shared__ __attribute__((aligned(32))) f16   xh[Bn * Ln];
    __shared__ __attribute__((aligned(32))) f16   xl[Bn * Ln];
    __shared__ __attribute__((aligned(32))) f16   Afh[WIN * 32 * 16];
    __shared__ __attribute__((aligned(32))) f16   Afl[WIN * 32 * 16];

    const int d    = blockIdx.x;
    const int tid  = threadIdx.x;
    const int lane = tid & 31;
    // wave id as an SGPR so per-tile lag caps compile to scalar branches
    // (EXEC must stay all-ones across v_wmma).
    const int w    = __builtin_amdgcn_readfirstlane(tid >> 5);

    // ---- stage filter into LDS: async global->LDS DMA path when available.
    // Builtin signature (from hipcc diagnostic): src is v4i in the device
    // (global) address space; dst is the LDS-side pointer; imm offset/cpol.
#if __has_builtin(__builtin_amdgcn_global_load_async_to_lds_b128)
    for (int e = tid * 4; e < Ln; e += BLOCK * 4) {
        __builtin_amdgcn_global_load_async_to_lds_b128(
            (AS1 v4i*)(void*)(fg + (size_t)d * Ln + e),
            (AS3 v4i*)(hbuf + e), 0, 0);
    }
    asm volatile("s_wait_asynccnt 0" ::: "memory");
#else
    for (int e = tid; e < Ln; e += BLOCK)
        hbuf[e] = fg[(size_t)d * Ln + e];
#endif

    // ---- stage x (all 4 batches) into LDS; split f32 -> f16 hi/lo
    for (int e = tid; e < Bn * Ln; e += BLOCK) {
        const int b = e >> 12, pos = e & (Ln - 1);
        const float f = xg[((size_t)b * Dn + d) * Ln + pos];
        const f16 hi = (f16)f;
        xh[e] = hi;
        xl[e] = (f16)(f - (float)hi);
    }
    __syncthreads();

    // Tiles handled by this wave: (b, T) for b=0..3, T in {w, 15-w}.
    // Work per tile ~ (T+1): pairing w with 15-w balances all waves exactly.
    const int T0 = w, T1 = 15 - w;
    v8f acc[8];
#pragma unroll
    for (int n = 0; n < 8; ++n) acc[n] = (v8f){};

    const int c = lane & 15;  // output-chunk column within tile (B/N layout)

    for (int stage = 0; stage < NSTAGES; ++stage) {
        // ---- cooperatively pre-gather filter Toeplitz A-fragments into LDS
        // in the contiguous-K A lane layout that B also uses:
        //   lane ln: row M = ln&15
        //   half j : K = j + (ln>=16 ? 16 : 0)
        //   K -> lag bit (K>>4) and in-chunk position q = K&15
        for (int e = tid; e < WIN * 512; e += BLOCK) {
            const int sl  = e >> 9;
            const int ln  = (e >> 4) & 31;
            const int j   = e & 15;
            const int kap = j + ((ln >= 16) ? 16 : 0);
            const int k   = 2 * (stage * WIN + sl) + (kap >> 4);
            const int q   = kap & 15;
            const int p   = ln & 15;
            const int idx = 16 * k + p - q;   // h[16k + p - q], causal zero-pad
            const float v = (idx >= 0 && idx < Ln) ? hbuf[idx] : 0.0f;
            const f16 hi = (f16)v;
            Afh[e] = hi;                       // e == sl*512 + ln*16 + j
            Afl[e] = (f16)(v - (float)hi);
        }
        __syncthreads();

#pragma unroll 2
        for (int sl = 0; sl < WIN; ++sl) {
            const int s = stage * WIN + sl;
            // A fragments: one contiguous 32B LDS read per lane, per part
            const v16h Ah = *(const v16h*)(Afh + sl * 512 + lane * 16);
            const v16h Al = *(const v16h*)(Afl + sl * 512 + lane * 16);
            // B layout: lanes 0-15 carry K=0..15 (lag 2s), lanes 16-31
            // carry K=16..31 (lag 2s+1); halves are q=0..15 -> one
            // contiguous x-chunk per lane.
            const int k = 2 * s + ((lane >= 16) ? 1 : 0);

#pragma unroll
            for (int n = 0; n < 8; ++n) {
                const int b = n >> 1;
                const int T = (n & 1) ? T1 : T0;
                const int i = 16 * T + c;
                if (s < 8 * T) {
                    // Off-diagonal block: every lane causally live (i > k
                    // for all lanes) -> no masking, pure load+WMMA.
                    const int off = 16 * (i - k);
                    const v16h Bh = *(const v16h*)(xh + b * Ln + off);
                    const v16h Bl = *(const v16h*)(xl + b * Ln + off);
                    acc[n] = __builtin_amdgcn_wmma_f32_16x16x32_f16(
                        false, Ah, false, Bh, (short)0, acc[n], false, false);
                    acc[n] = __builtin_amdgcn_wmma_f32_16x16x32_f16(
                        false, Ah, false, Bl, (short)0, acc[n], false, false);
                    acc[n] = __builtin_amdgcn_wmma_f32_16x16x32_f16(
                        false, Al, false, Bh, (short)0, acc[n], false, false);
                } else if (s <= 8 * T + 7) {
                    // Diagonal block (last 8 steps): per-lane causal mask via
                    // clamped address + v_cndmask select, no divergent branch.
                    const bool live = (i >= k);
                    const int  off  = live ? 16 * (i - k) : 0;
                    const v16h th = *(const v16h*)(xh + b * Ln + off);
                    const v16h tl = *(const v16h*)(xl + b * Ln + off);
                    const v16h z  = {};
                    const v16h Bh = live ? th : z;
                    const v16h Bl = live ? tl : z;
                    acc[n] = __builtin_amdgcn_wmma_f32_16x16x32_f16(
                        false, Ah, false, Bh, (short)0, acc[n], false, false);
                    acc[n] = __builtin_amdgcn_wmma_f32_16x16x32_f16(
                        false, Ah, false, Bl, (short)0, acc[n], false, false);
                    acc[n] = __builtin_amdgcn_wmma_f32_16x16x32_f16(
                        false, Al, false, Bh, (short)0, acc[n], false, false);
                }
            }
        }
        __syncthreads();  // A-frag buffers are rewritten next stage
    }

    // ---- write out. C/D layout: VGPR r -> row p=r (+8 for lanes 16-31),
    // lane&15 -> column chunk c. Each lane's 8 results are CONTIGUOUS in y
    // (base+prow .. base+prow+7, 32B-aligned) -> one v8f (2x b128) store.
    const int prow = (lane >= 16) ? 8 : 0;
#pragma unroll
    for (int n = 0; n < 8; ++n) {
        const int b = n >> 1;
        const int T = (n & 1) ? T1 : T0;
        const size_t base = ((size_t)b * Dn + d) * Ln + 16 * (16 * T + c);
        *(v8f*)(yg + base + prow) = acc[n];
    }
}

extern "C" void kernel_launch(void* const* d_in, const int* in_sizes, int n_in,
                              void* d_out, int out_size, void* d_ws, size_t ws_size,
                              hipStream_t stream) {
    (void)in_sizes; (void)n_in; (void)d_ws; (void)ws_size; (void)out_size;
    const float* x    = (const float*)d_in[0];
    const float* filt = (const float*)d_in[1];
    float*       y    = (float*)d_out;
    fftconv_lag_wmma<<<dim3(Dn), dim3(BLOCK), 0, stream>>>(x, filt, y);
}
